// PointNetFeaturePropagation_30270929502681
// MI455X (gfx1250) — matrix operand order repaired and checked by reading, not compiled, not run on hardware
//
#include <hip/hip_runtime.h>
#include <hip/hip_bf16.h>

#define B_   4
#define N1_  16384
#define N2_  4096
#define C1_  128
#define C2_  256
#define CIN_ 384
#define H1_  256
#define H2_  128

typedef __attribute__((ext_vector_type(16))) __bf16 v16bf;
typedef __attribute__((ext_vector_type(8)))  float  v8f;

union Frag { uint4 q[2]; v16bf v; };

__device__ __forceinline__ unsigned int pk_bf16(float lo, float hi) {
    unsigned short l = __builtin_bit_cast(unsigned short, (__bf16)lo);
    unsigned short h = __builtin_bit_cast(unsigned short, (__bf16)hi);
    return (unsigned int)l | ((unsigned int)h << 16);
}

// ---------------------------------------------------------------------------
// Repack W (row-major KxN f32) into per-lane WMMA B-fragment order, bf16.
// dword index = ((nt*Ksteps + ks)*32 + lane)*8 + j
// lane<16 -> K offset 0, lane>=16 -> K offset 8; dword j holds K=16*(j>>2)+2*(j&3),+1.
// ---------------------------------------------------------------------------
__global__ void repack_kernel(const float* __restrict__ W,
                              unsigned int* __restrict__ dst,
                              int Ksteps, int Ntiles, int N) {
    int idx = blockIdx.x * blockDim.x + threadIdx.x;
    int total = Ntiles * Ksteps * 256;
    if (idx >= total) return;
    int j    = idx & 7;
    int lane = (idx >> 3) & 31;
    int t    = idx >> 8;
    int ks   = t % Ksteps;
    int nt   = t / Ksteps;
    int k    = ks * 32 + ((lane >> 4) << 3) + ((j >> 2) << 4) + 2 * (j & 3);
    int n    = nt * 16 + (lane & 15);
    dst[idx] = pk_bf16(W[k * N + n], W[(k + 1) * N + n]);
}

// ---------------------------------------------------------------------------
// 3-NN + inverse-distance interpolation + concat -> X [B*N1, 384] bf16
// ---------------------------------------------------------------------------
__global__ __launch_bounds__(256)
void knn_interp_kernel(const float* __restrict__ xyz1,
                       const float* __restrict__ xyz2,
                       const float* __restrict__ f1,
                       const float* __restrict__ f2,
                       __bf16* __restrict__ X) {
    __shared__ float sx[N2_ * 3];                 // 48 KB
    const int tid = threadIdx.x;
    const int b   = blockIdx.y;
    for (int i = tid; i < N2_ * 3; i += 256)
        sx[i] = xyz2[(long long)b * N2_ * 3 + i];
    __syncthreads();

    const int n = blockIdx.x * 256 + tid;
    const long long row = (long long)b * N1_ + n;
    const float qx = xyz1[row * 3 + 0];
    const float qy = xyz1[row * 3 + 1];
    const float qz = xyz1[row * 3 + 2];

    float d0 = 1e30f, d1 = 1e30f, d2 = 1e30f;
    int   i0 = 0,     i1 = 0,     i2 = 0;
    for (int j = 0; j < N2_; ++j) {
        float dx = qx - sx[3 * j + 0];
        float dy = qy - sx[3 * j + 1];
        float dz = qz - sx[3 * j + 2];
        float d  = dx * dx + dy * dy + dz * dz;
        if (d < d2) {
            if (d < d1) {
                if (d < d0) { d2 = d1; i2 = i1; d1 = d0; i1 = i0; d0 = d; i0 = j; }
                else        { d2 = d1; i2 = i1; d1 = d;  i1 = j; }
            } else          { d2 = d;  i2 = j; }
        }
    }
    float w0 = 1.0f / fmaxf(d0, 1e-10f);
    float w1 = 1.0f / fmaxf(d1, 1e-10f);
    float w2 = 1.0f / fmaxf(d2, 1e-10f);
    float ws = 1.0f / (w0 + w1 + w2);
    w0 *= ws; w1 *= ws; w2 *= ws;

    const float4* g0 = (const float4*)(f2 + ((long long)b * N2_ + i0) * C2_);
    const float4* g1 = (const float4*)(f2 + ((long long)b * N2_ + i1) * C2_);
    const float4* g2 = (const float4*)(f2 + ((long long)b * N2_ + i2) * C2_);
    __bf16* xr = X + row * CIN_;

    #pragma unroll 4
    for (int c = 0; c < C2_ / 4; ++c) {
        float4 a = g0[c], bb = g1[c], cc = g2[c];
        float v0 = w0 * a.x + w1 * bb.x + w2 * cc.x;
        float v1 = w0 * a.y + w1 * bb.y + w2 * cc.y;
        float v2 = w0 * a.z + w1 * bb.z + w2 * cc.z;
        float v3 = w0 * a.w + w1 * bb.w + w2 * cc.w;
        uint2 u; u.x = pk_bf16(v0, v1); u.y = pk_bf16(v2, v3);
        *(uint2*)(xr + 4 * c) = u;
    }
    const float4* s1 = (const float4*)(f1 + row * C1_);
    #pragma unroll 4
    for (int c = 0; c < C1_ / 4; ++c) {
        float4 a = s1[c];
        uint2 u; u.x = pk_bf16(a.x, a.y); u.y = pk_bf16(a.z, a.w);
        *(uint2*)(xr + C2_ + 4 * c) = u;
    }
}

// ---------------------------------------------------------------------------
// Fused two-layer pointwise MLP via bf16 WMMA.
// 8 waves / block, each wave owns 16 rows; H kept in wave-private LDS in
// GEMM2 A-fragment order (8 KB/wave, 64 KB total, no barriers needed).
// ---------------------------------------------------------------------------
__global__ __launch_bounds__(256)
void mlp_kernel(const __bf16* __restrict__ X,
                const unsigned int* __restrict__ W1p,
                const unsigned int* __restrict__ W2p,
                const float* __restrict__ b1,
                const float* __restrict__ b2,
                float* __restrict__ out) {
    __shared__ __align__(16) unsigned char sFrag[8][8192];   // 64 KB
    const int tid  = threadIdx.x;
    const int lane = tid & 31;
    const int wave = tid >> 5;
    const int ln   = lane & 15;
    const int hi   = lane >> 4;          // 0 or 1
    const int sel  = hi * 8;             // K element offset for A fragments
    const long long mbase = (long long)blockIdx.x * 128 + wave * 16;
    unsigned char* frag = sFrag[wave];
    const __bf16* Arow = X + (mbase + ln) * CIN_;

    // -------- GEMM1: H = relu(X @ W1 + b1), N=256 in two halves of 8 tiles
    for (int half = 0; half < 2; ++half) {
        v8f acc[8] = {};
        #pragma unroll
        for (int ks = 0; ks < 12; ++ks) {
            Frag a;
            a.q[0] = *(const uint4*)(Arow + ks * 32 + sel);
            a.q[1] = *(const uint4*)(Arow + ks * 32 + sel + 16);
            #pragma unroll
            for (int t = 0; t < 8; ++t) {
                int nt = half * 8 + t;
                const uint4* bp = (const uint4*)(W1p + (((nt * 12 + ks) * 32 + lane) * 8));
                Frag bm; bm.q[0] = bp[0]; bm.q[1] = bp[1];
                acc[t] = __builtin_amdgcn_wmma_f32_16x16x32_bf16(
                    false, a.v, false, bm.v, (short)0, acc[t], false, false);
            }
        }
        // epilogue: bias + relu -> bf16, scatter into GEMM2 A-fragment layout
        #pragma unroll
        for (int t = 0; t < 8; ++t) {
            int nt = half * 8 + t;
            int n  = nt * 16 + ln;
            float bv = b1[n];
            int ks2  = n >> 5;
            int r5   = n & 31;
            int part = (r5 >> 4) & 1;
            int b8   = (r5 >> 3) & 1;
            unsigned int base = (unsigned)(ks2 * 1024 + part * 512 + b8 * 256 +
                                           hi * 128 + (r5 & 7) * 2);
            #pragma unroll
            for (int r = 0; r < 8; ++r) {
                float h = fmaxf(acc[t][r] + bv, 0.0f);
                *(__bf16*)(frag + base + r * 16) = (__bf16)h;
            }
        }
    }
    asm volatile("s_wait_dscnt 0" ::: "memory");   // wave-private RAW on LDS

    // -------- GEMM2: out = relu(H @ W2 + b2), N=128
    v8f acc2[8] = {};
    #pragma unroll
    for (int ks = 0; ks < 8; ++ks) {
        Frag a;
        a.q[0] = *(const uint4*)(frag + ks * 1024 + lane * 16);
        a.q[1] = *(const uint4*)(frag + ks * 1024 + 512 + lane * 16);
        #pragma unroll
        for (int nt = 0; nt < 8; ++nt) {
            const uint4* bp = (const uint4*)(W2p + (((nt * 8 + ks) * 32 + lane) * 8));
            Frag bm; bm.q[0] = bp[0]; bm.q[1] = bp[1];
            acc2[nt] = __builtin_amdgcn_wmma_f32_16x16x32_bf16(
                false, a.v, false, bm.v, (short)0, acc2[nt], false, false);
        }
    }
    #pragma unroll
    for (int nt = 0; nt < 8; ++nt) {
        int n = nt * 16 + ln;
        float bv = b2[n];
        #pragma unroll
        for (int r = 0; r < 8; ++r) {
            int m = r + 8 * hi;
            out[(mbase + m) * H2_ + n] = fmaxf(acc2[nt][r] + bv, 0.0f);
        }
    }
}

// ---------------------------------------------------------------------------
extern "C" void kernel_launch(void* const* d_in, const int* in_sizes, int n_in,
                              void* d_out, int out_size, void* d_ws, size_t ws_size,
                              hipStream_t stream) {
    const float* xyz1 = (const float*)d_in[0];
    const float* xyz2 = (const float*)d_in[1];
    const float* f1   = (const float*)d_in[2];
    const float* f2   = (const float*)d_in[3];
    const float* W1   = (const float*)d_in[4];
    const float* b1   = (const float*)d_in[5];
    const float* W2   = (const float*)d_in[6];
    const float* b2   = (const float*)d_in[7];
    float* out = (float*)d_out;

    // workspace layout
    unsigned char* ws = (unsigned char*)d_ws;
    const size_t xBytes  = (size_t)B_ * N1_ * CIN_ * 2;        // 48 MB bf16 X
    const size_t w1Bytes = (size_t)16 * 12 * 32 * 8 * 4;       // 192 KB
    __bf16*       Xbf = reinterpret_cast<__bf16*>(ws);
    unsigned int* W1p = reinterpret_cast<unsigned int*>(ws + xBytes);
    unsigned int* W2p = reinterpret_cast<unsigned int*>(ws + xBytes + w1Bytes);

    // 1) repack weights into WMMA B-fragment order (bf16)
    repack_kernel<<<(16 * 12 * 256 + 255) / 256, 256, 0, stream>>>(W1, W1p, 12, 16, H1_);
    repack_kernel<<<(8 * 8 * 256 + 255) / 256, 256, 0, stream>>>(W2, W2p, 8, 8, H2_);

    // 2) kNN + interpolation + concat -> X
    knn_interp_kernel<<<dim3(N1_ / 256, B_), 256, 0, stream>>>(xyz1, xyz2, f1, f2, Xbf);

    // 3) fused MLP (two WMMA GEMMs)
    mlp_kernel<<<(B_ * N1_) / 128, 256, 0, stream>>>(Xbf, W1p, W2p, b1, b2, out);
}
